// MyModel_61933428411464
// MI455X (gfx1250) — compile-verified, matrix operand order checked by reading, not run on hardware
//
#include <hip/hip_runtime.h>

typedef __attribute__((ext_vector_type(16))) _Float16 v16h;
typedef __attribute__((ext_vector_type(8)))  float    v8f;

#define TSEQ 2048
#define HIDN 5

__device__ __forceinline__ float frcp(float x)  { return __builtin_amdgcn_rcpf(x); }
__device__ __forceinline__ float fexp2(float x) { return __builtin_amdgcn_exp2f(x); }

constexpr float L2E = 1.4426950408889634f;

// sigmoid(z) = 1/(1+2^(-L2E z));  tanh(z) = 2*sigmoid(2z)-1
__device__ __forceinline__ float sigm(float z) {
  return frcp(1.f + fexp2(-L2E * z));
}

// -------- pass 1: ragged lengths = count of nonzero timesteps per row --------
__global__ void lengths_kernel(const float* __restrict__ x,
                               int* __restrict__ lengths, int B) {
  int gid  = blockIdx.x * blockDim.x + threadIdx.x;
  int row  = gid >> 5;
  int lane = threadIdx.x & 31;
  if (row >= B) return;
  const float* xr = x + (size_t)row * TSEQ;
  int cnt = 0;
  for (int t = lane * 4; t < TSEQ; t += 128) {
    float4 v = *(const float4*)(xr + t);
    cnt += (v.x != 0.f) + (v.y != 0.f) + (v.z != 0.f) + (v.w != 0.f);
  }
  #pragma unroll
  for (int off = 16; off > 0; off >>= 1) cnt += __shfl_xor(cnt, off, 32);
  if (lane == 0) lengths[row] = cnt;
}

// -------- pass 2: WMMA LSTM recurrence, 16 samples per wave --------
__global__ void __launch_bounds__(128)
lstm_wmma_kernel(const float* __restrict__ x,
                 const float* __restrict__ Wih,
                 const float* __restrict__ Whh,
                 const float* __restrict__ bih,
                 const float* __restrict__ bhh,
                 const int*   __restrict__ lengths,
                 float* __restrict__ out, int B) {
  const int lane = threadIdx.x & 31;
  const int wave = threadIdx.x >> 5;
  const int s    = lane & 15;       // sample slot within the wave's 16
  const int hi   = lane >> 4;       // 0 = lower half, 1 = upper half
  const bool lo  = (hi == 0);
  const int sampleBase = (blockIdx.x * 4 + wave) * 16;
  if (sampleBase >= B) return;

  // ---- constant A tiles: W_aug[m][k], m=gate, k=[h0..h4, x] ----
  v16h a1; v16h a2;
  #pragma unroll
  for (int k = 0; k < 16; ++k) { a1[k] = (_Float16)0.f; a2[k] = (_Float16)0.f; }
  if (lo) {
    #pragma unroll
    for (int k = 0; k < HIDN; ++k) a1[k] = (_Float16)Whh[s * HIDN + k];
    a1[5] = (_Float16)Wih[s];
    if (s < 4) {
      #pragma unroll
      for (int k = 0; k < HIDN; ++k) a2[k] = (_Float16)Whh[(16 + s) * HIDN + k];
      a2[5] = (_Float16)Wih[16 + s];
    }
  }

  // ---- constant bias in C/D layout: D[m=gate][n=sample] ----
  v8f cb1, cb2;
  #pragma unroll
  for (int v = 0; v < 8; ++v) {
    int g = v + 8 * hi;
    cb1[v] = bih[g] + bhh[g];
    cb2[v] = (g < 4) ? (bih[16 + g] + bhh[16 + g]) : 0.f;  // gates 16..19
  }

  // D1 rows 2..6: lower lanes = gates 2..6 (sigmoid), upper = gates 10..14 (tanh)
  const float km = lo ? -L2E : -2.f * L2E;
  const float am = lo ? 1.f  : 2.f;
  const float bm = lo ? 0.f  : -1.f;

  int len = lo ? lengths[sampleBase + s] : 0;   // upper half frozen forever
  int maxlen = len;
  #pragma unroll
  for (int off = 16; off > 0; off >>= 1) {
    int o = __shfl_xor(maxlen, off, 32);
    maxlen = maxlen > o ? maxlen : o;
  }

  const float* xr = x + (size_t)(sampleBase + s) * TSEQ;

  float h0=0,h1=0,h2=0,h3=0,h4=0;
  float c0=0,c1=0,c2=0,c3=0,c4=0;

  for (int t0 = 0; t0 < maxlen; t0 += 4) {
    if ((t0 & 127) == 0) __builtin_prefetch(xr + t0 + 512, 0, 1);
    float4 xq = *(const float4*)(xr + t0);
    float xs4[4] = {xq.x, xq.y, xq.z, xq.w};
    #pragma unroll
    for (int u = 0; u < 4; ++u) {
      int t = t0 + u;
      if (t >= maxlen) break;               // uniform across wave
      float xt = lo ? xs4[u] : 0.f;

      // B operand: lane n holds [h0..h4, x] of its sample; K>=6 and upper half = 0
      v16h b;
      #pragma unroll
      for (int k = 6; k < 16; ++k) b[k] = (_Float16)0.f;
      b[0]=(_Float16)h0; b[1]=(_Float16)h1; b[2]=(_Float16)h2;
      b[3]=(_Float16)h3; b[4]=(_Float16)h4; b[5]=(_Float16)xt;

      v8f d1 = __builtin_amdgcn_wmma_f32_16x16x32_f16(false, a1, false, b,
                                                      (short)0, cb1, false, false);
      v8f d2 = __builtin_amdgcn_wmma_f32_16x16x32_f16(false, a2, false, b,
                                                      (short)0, cb2, false, false);

      // nonlinearities in D layout (all 32 lanes busy on d1)
      float n0 = sigm(d1[0]);
      float n1 = sigm(d1[1]);
      float n2 = am * frcp(1.f + fexp2(km * d1[2])) + bm;
      float n3 = am * frcp(1.f + fexp2(km * d1[3])) + bm;
      float n4 = am * frcp(1.f + fexp2(km * d1[4])) + bm;
      float n5 = am * frcp(1.f + fexp2(km * d1[5])) + bm;
      float n6 = am * frcp(1.f + fexp2(km * d1[6])) + bm;
      float n7 = sigm(d1[7]);
      float o1 = sigm(d2[0]);
      float o2 = sigm(d2[1]);
      float o3 = sigm(d2[2]);
      float o4 = sigm(d2[3]);

      // bring the upper-half gates (8..15) of sample s down to lane s
      float f3 = __shfl_xor(n0, 16, 32);
      float f4 = __shfl_xor(n1, 16, 32);
      float g0 = __shfl_xor(n2, 16, 32);
      float g1 = __shfl_xor(n3, 16, 32);
      float g2 = __shfl_xor(n4, 16, 32);
      float g3 = __shfl_xor(n5, 16, 32);
      float g4 = __shfl_xor(n6, 16, 32);
      float o0 = __shfl_xor(n7, 16, 32);
      // lane-local on lower half: i0..4 = n0..n4, f0..2 = n5..n7

      bool upd = (t < len);                 // false on upper half & finished lanes
      float cn, th, hn;
      cn = n5 * c0 + n0 * g0; th = 2.f*frcp(1.f + fexp2(-2.f*L2E*cn)) - 1.f; hn = o0*th;
      c0 = upd ? cn : c0;  h0 = upd ? hn : h0;
      cn = n6 * c1 + n1 * g1; th = 2.f*frcp(1.f + fexp2(-2.f*L2E*cn)) - 1.f; hn = o1*th;
      c1 = upd ? cn : c1;  h1 = upd ? hn : h1;
      cn = n7 * c2 + n2 * g2; th = 2.f*frcp(1.f + fexp2(-2.f*L2E*cn)) - 1.f; hn = o2*th;
      c2 = upd ? cn : c2;  h2 = upd ? hn : h2;
      cn = f3 * c3 + n3 * g3; th = 2.f*frcp(1.f + fexp2(-2.f*L2E*cn)) - 1.f; hn = o3*th;
      c3 = upd ? cn : c3;  h3 = upd ? hn : h3;
      cn = f4 * c4 + n4 * g4; th = 2.f*frcp(1.f + fexp2(-2.f*L2E*cn)) - 1.f; hn = o4*th;
      c4 = upd ? cn : c4;  h4 = upd ? hn : h4;
    }
  }

  if (lo) {
    float* op = out + (size_t)(sampleBase + s) * HIDN;
    op[0]=h0; op[1]=h1; op[2]=h2; op[3]=h3; op[4]=h4;
  }
}

extern "C" void kernel_launch(void* const* d_in, const int* in_sizes, int n_in,
                              void* d_out, int out_size, void* d_ws, size_t ws_size,
                              hipStream_t stream) {
  const float* x   = (const float*)d_in[0];
  const float* Wih = (const float*)d_in[1];
  const float* Whh = (const float*)d_in[2];
  const float* bih = (const float*)d_in[3];
  const float* bhh = (const float*)d_in[4];
  float* out = (float*)d_out;
  int B = in_sizes[0] / TSEQ;

  int* lengths = (int*)d_ws;

  int blocks1 = (B * 32 + 255) / 256;          // one wave per row
  lengths_kernel<<<blocks1, 256, 0, stream>>>(x, lengths, B);

  int blocks2 = (B + 63) / 64;                 // 4 waves/block * 16 samples/wave
  lstm_wmma_kernel<<<blocks2, 128, 0, stream>>>(x, Wih, Whh, bih, bhh, lengths, out, B);
}
